// RotaryGQA_42228118454586
// MI455X (gfx1250) — matrix operand order repaired and checked by reading, not compiled, not run on hardware
//
#include <hip/hip_runtime.h>
#include <hip/hip_bf16.h>

// ---------------------------------------------------------------------------
// GQA attention block for MI455X (gfx1250, wave32, WMMA + TDM).
// Pipeline: f32->f16 convert, WMMA GEMM (QKV proj, TDM-staged A), RoPE,
// WMMA flash attention (TDM-staged K/V chunks, ds_swizzle softmax butterfly),
// WMMA GEMM (out proj, f32).
// ---------------------------------------------------------------------------

typedef __attribute__((ext_vector_type(16))) _Float16     v16h;
typedef __attribute__((ext_vector_type(8)))  float        v8f;
typedef __attribute__((ext_vector_type(4)))  unsigned int v4u;
typedef __attribute__((ext_vector_type(8)))  unsigned int v8u;

#define B_  2
#define T_  2048
#define C_  1024
#define NH_ 16
#define NKV_ 4
#define HD_ 64
#define M_  (B_ * T_)        // 4096 rows
#define KVC_ (NKV_ * HD_)    // 256

// A-fragment (16x32 f16, MxK) per-lane K index for half element e (ISA 7.12.2)
__device__ __forceinline__ int akk(int e, int hi) {
  return (e & 7) + 8 * hi + ((e >> 3) << 4);
}
// B-fragment (32x16 f16, KxN) per-lane K index (same table order as A)
__device__ __forceinline__ int bkk(int e, int hi) {
  return (e & 7) + 8 * hi + ((e >> 3) << 4);
}

// XOR-butterfly lane exchange via ds_swizzle group-of-32 mode:
// offset = {0, xor[14:10], or[9:5]=0, and[4:0]=0x1F}
template <int XMASK>
__device__ __forceinline__ float swz_xor(float v) {
  return __int_as_float(
      __builtin_amdgcn_ds_swizzle(__float_as_int(v), (XMASK << 10) | 0x1F));
}

// ---------------------------------------------------------------------------
// Tensor Data Mover: 2D tile (16-bit elements) global -> LDS, with optional
// per-row LDS padding. Descriptor per CDNA5 ISA.
// Issue from ONE wave; that wave must s_wait_tensorcnt before the barrier.
// ---------------------------------------------------------------------------
__device__ __forceinline__ void tdm_load_2d_f16(
    unsigned lds_addr, const void* gptr,
    unsigned tile_d0, unsigned tile_d1,            // elements
    unsigned long long stride0_elems,              // row stride, elements
    unsigned pad_interval_code, unsigned pad_amount_code, unsigned pad_enable) {
  unsigned long long ga = (unsigned long long)gptr;
  v4u g0;
  g0[0] = 1u;                                        // count=1, user mode
  g0[1] = lds_addr;                                  // LDS byte address
  g0[2] = (unsigned)ga;                              // global addr [31:0]
  g0[3] = (unsigned)((ga >> 32) & 0x01FFFFFFu)       // global addr [56:32]
          | (2u << 30);                              // type = 2 ("image")
  v8u g1;
  g1[0] = (1u << 16)                                 // data_size: 2 bytes
          | (pad_enable << 20)
          | (pad_interval_code << 22)
          | (pad_amount_code << 25);
  g1[1] = (tile_d0 & 0xFFFFu) << 16;                 // tensor_dim0[15:0]
  g1[2] = ((tile_d0 >> 16) & 0xFFFFu)                // tensor_dim0[31:16]
          | ((tile_d1 & 0xFFFFu) << 16);             // tensor_dim1[15:0]
  g1[3] = ((tile_d1 >> 16) & 0xFFFFu)                // tensor_dim1[31:16]
          | ((tile_d0 & 0xFFFFu) << 16);             // tile_dim0
  g1[4] = (tile_d1 & 0xFFFFu);                       // tile_dim1; tile_dim2=0
  g1[5] = (unsigned)(stride0_elems & 0xFFFFFFFFull); // dim0_stride[31:0]
  g1[6] = (unsigned)((stride0_elems >> 32) & 0xFFFFull); // dim0_stride[47:32]
  g1[7] = 0u;                                        // dim1_stride unused (2D)
  asm volatile("tensor_load_to_lds %0, %1" : : "s"(g0), "s"(g1) : "memory");
}

// ---------------------------------------------------------------------------
// f32 -> f16 conversion
// ---------------------------------------------------------------------------
__global__ void cvt_f32_to_f16(const float* __restrict__ x,
                               _Float16* __restrict__ y, int n) {
  int i = blockIdx.x * blockDim.x + threadIdx.x;
  if (i < n) y[i] = (_Float16)x[i];
}

// ---------------------------------------------------------------------------
// GEMM: out[M,N] = Xh[M,K] (f16) @ W[K,N] (f32->f16) + bias
// Block: 256 threads = 8 waves, tile 256(M) x 64(N), k-step 32.
// A tile staged by TDM (wave 0); W tile staged/transposed by vector loads.
// Each wave computes a 32x64 strip: 8 WMMA per k-step.
// ---------------------------------------------------------------------------
template <bool F32OUT>
__global__ __launch_bounds__(256) void gemm_wmma(
    const _Float16* __restrict__ X, const float* __restrict__ W,
    const float* __restrict__ bias, void* __restrict__ outp,
    int M, int N, int K) {
  __shared__ _Float16 sA[256][34];  // 32 data halves + 2 pad (1 DWORD)
  __shared__ _Float16 sB[64][34];   // B transposed: sB[n][k]

  const int tid = threadIdx.x;
  const int w = tid >> 5, lane = tid & 31, l16 = lane & 15, hi = lane >> 4;
  const int rowBase = blockIdx.y * 256;
  const int n0 = blockIdx.x * 64;
  const unsigned ldsA = (unsigned)(size_t)&sA[0][0];

  v8f acc0[4] = {};
  v8f acc1[4] = {};

  for (int k0 = 0; k0 < K; k0 += 32) {
    __syncthreads();
    if (tid < 32) {  // wave 0: DMA the 256x32 f16 A tile, rows padded +1 DWORD
      tdm_load_2d_f16(ldsA, X + (size_t)rowBase * K + k0, 32u, 256u,
                      (unsigned long long)K, /*interval 64B*/ 3u,
                      /*pad 1 DWORD*/ 0u, 1u);
      __builtin_amdgcn_s_wait_tensorcnt(0);
    }
    {  // stage B (transposed): 32x64 f32 -> f16, 8 per thread
      int bk = tid >> 3;
      int bn = (tid & 7) * 8;
      const float* wp = W + (size_t)(k0 + bk) * N + n0 + bn;
#pragma unroll
      for (int j = 0; j < 8; ++j) sB[bn + j][bk] = (_Float16)wp[j];
      if (k0 + 32 < K) __builtin_prefetch(wp + (size_t)32 * N, 0, 1);
    }
    __syncthreads();

    v16h a0, a1;
#pragma unroll
    for (int e = 0; e < 16; ++e) {
      int kd = akk(e, hi);
      a0[e] = sA[32 * w + l16][kd];
      a1[e] = sA[32 * w + 16 + l16][kd];
    }
    v16h bb[4];
#pragma unroll
    for (int j = 0; j < 4; ++j)
#pragma unroll
      for (int e = 0; e < 16; ++e) bb[j][e] = sB[16 * j + l16][bkk(e, hi)];
#pragma unroll
    for (int j = 0; j < 4; ++j)
      acc0[j] = __builtin_amdgcn_wmma_f32_16x16x32_f16(
          false, a0, false, bb[j], (short)0, acc0[j], false, false);
#pragma unroll
    for (int j = 0; j < 4; ++j)
      acc1[j] = __builtin_amdgcn_wmma_f32_16x16x32_f16(
          false, a1, false, bb[j], (short)0, acc1[j], false, false);
  }

#pragma unroll
  for (int j = 0; j < 4; ++j) {
    int col = n0 + 16 * j + l16;
    float bv = bias[col];
#pragma unroll
    for (int r = 0; r < 8; ++r) {  // C/D layout: m = r + 8*hi
      int row0 = rowBase + 32 * w + 8 * hi + r;
      int row1 = row0 + 16;
      float v0 = acc0[j][r] + bv;
      float v1 = acc1[j][r] + bv;
      if (F32OUT) {
        ((float*)outp)[(size_t)row0 * N + col] = v0;
        ((float*)outp)[(size_t)row1 * N + col] = v1;
      } else {
        ((_Float16*)outp)[(size_t)row0 * N + col] = (_Float16)v0;
        ((_Float16*)outp)[(size_t)row1 * N + col] = (_Float16)v1;
      }
    }
  }
}

// ---------------------------------------------------------------------------
// RoPE, interleaved pairs (2i, 2i+1), in-place on f16 [B*T, nH, 64]
// ---------------------------------------------------------------------------
__global__ void rope_kernel(_Float16* __restrict__ t, int nH, int total) {
  int idx = blockIdx.x * blockDim.x + threadIdx.x;
  if (idx >= total) return;
  int pair = idx & 31;
  int tmp = idx >> 5;
  int head = tmp % nH;
  int row = tmp / nH;          // row in [0, B*T)
  int tpos = row % T_;
  // inv_freq = 10000^(-2*pair/64); ln(10000)=9.2103403719761836
  float inv_freq = __expf(-(float)(2 * pair) * (9.2103403719761836f / 64.0f));
  float theta = (float)tpos * inv_freq;
  float c = __cosf(theta), s = __sinf(theta);
  size_t base = ((size_t)row * nH + head) * 64 + 2 * pair;
  float e = (float)t[base];
  float o = (float)t[base + 1];
  t[base]     = (_Float16)(e * c - o * s);
  t[base + 1] = (_Float16)(e * s + o * c);
}

// ---------------------------------------------------------------------------
// Flash attention (causal, GQA). Block = 256 threads (8 waves) handles 128
// query rows of one (b, h). Each wave owns 16 q rows. 32-key chunks are
// DMA-staged into LDS by TDM (wave 0); online softmax in f32 with
// ds_swizzle XOR butterflies; both matmuls via WMMA.
// ---------------------------------------------------------------------------
__global__ __launch_bounds__(256) void flash_attn_kernel(
    const _Float16* __restrict__ qg, const _Float16* __restrict__ kg,
    const _Float16* __restrict__ vg, _Float16* __restrict__ og) {
  __shared__ _Float16 sK[32][HD_ + 4];  // +4 halves (2 DWORDs) pad per row
  __shared__ _Float16 sV[32][HD_ + 4];
  __shared__ _Float16 sP[8][16][34];    // per-wave P staging (C- -> A-layout)

  const int tid = threadIdx.x;
  const int w = tid >> 5, lane = tid & 31, l16 = lane & 15, hi = lane >> 4;
  const int qbase = blockIdx.x * 128;
  const int h = blockIdx.y, b = blockIdx.z;
  const int kv = h >> 2;  // repeat_interleave: head h uses kv head h/4
  const int qrow = qbase + 16 * w + l16;
  const unsigned ldsK = (unsigned)(size_t)&sK[0][0];
  const unsigned ldsV = (unsigned)(size_t)&sV[0][0];

  // Q fragments for d-chunks [0,32) and [32,64), resident for whole block
  v16h aq0, aq1;
  {
    const _Float16* qp = qg + (((size_t)b * T_ + qrow) * NH_ + h) * HD_;
#pragma unroll
    for (int e = 0; e < 16; ++e) {
      int kd = akk(e, hi);
      aq0[e] = qp[kd];
      aq1[e] = qp[32 + kd];
    }
  }

  v8f oacc[4] = {};
  float mrow[8], lrow[8];
#pragma unroll
  for (int r = 0; r < 8; ++r) { mrow[r] = -1e30f; lrow[r] = 0.0f; }

  const int kend = qbase + 128;  // block-uniform trip count (causal bound)
  for (int kb = 0; kb < kend; kb += 32) {
    __syncthreads();
    if (tid < 32) {  // wave 0: DMA 32x64 f16 K and V chunks (row stride 256)
      const void* kp = kg + (((size_t)b * T_ + kb) * NKV_ + kv) * HD_;
      const void* vp = vg + (((size_t)b * T_ + kb) * NKV_ + kv) * HD_;
      tdm_load_2d_f16(ldsK, kp, 64u, 32u, (unsigned long long)(NKV_ * HD_),
                      /*interval 128B*/ 4u, /*pad 2 DWORDs*/ 1u, 1u);
      tdm_load_2d_f16(ldsV, vp, 64u, 32u, (unsigned long long)(NKV_ * HD_),
                      4u, 1u, 1u);
      __builtin_amdgcn_s_wait_tensorcnt(0);
    }
    __syncthreads();

    // S = Q * K^T : two 16-key subtiles, each accumulates over d in 2 WMMA
    v8f s[2] = {};
#pragma unroll
    for (int st = 0; st < 2; ++st) {
      v16h b0, b1;
#pragma unroll
      for (int e = 0; e < 16; ++e) {
        int kd = bkk(e, hi);
        b0[e] = sK[st * 16 + l16][kd];
        b1[e] = sK[st * 16 + l16][32 + kd];
      }
      s[st] = __builtin_amdgcn_wmma_f32_16x16x32_f16(
          false, aq0, false, b0, (short)0, s[st], false, false);
      s[st] = __builtin_amdgcn_wmma_f32_16x16x32_f16(
          false, aq1, false, b1, (short)0, s[st], false, false);
    }

    // scale + causal mask + row max
    float rmax[8];
#pragma unroll
    for (int r = 0; r < 8; ++r) rmax[r] = -1e30f;
#pragma unroll
    for (int st = 0; st < 2; ++st) {
      int key = kb + st * 16 + l16;
#pragma unroll
      for (int r = 0; r < 8; ++r) {
        int qr = qbase + 16 * w + 8 * hi + r;
        float vsc = s[st][r] * 0.125f;  // 1/sqrt(64)
        if (key > qr) vsc = -1e30f;
        s[st][r] = vsc;
        rmax[r] = fmaxf(rmax[r], vsc);
      }
    }
    // XOR butterfly over the 16 lanes holding one row (ds_swizzle, no addr)
#pragma unroll
    for (int r = 0; r < 8; ++r) {
      rmax[r] = fmaxf(rmax[r], swz_xor<1>(rmax[r]));
      rmax[r] = fmaxf(rmax[r], swz_xor<2>(rmax[r]));
      rmax[r] = fmaxf(rmax[r], swz_xor<4>(rmax[r]));
      rmax[r] = fmaxf(rmax[r], swz_xor<8>(rmax[r]));
    }

    float alpha[8], rsum[8];
#pragma unroll
    for (int r = 0; r < 8; ++r) {
      float mn = fmaxf(mrow[r], rmax[r]);
      alpha[r] = __expf(mrow[r] - mn);
      mrow[r] = mn;
      rsum[r] = 0.0f;
    }
#pragma unroll
    for (int st = 0; st < 2; ++st)
#pragma unroll
      for (int r = 0; r < 8; ++r) {
        float p = __expf(s[st][r] - mrow[r]);
        s[st][r] = p;
        rsum[r] += p;
      }
#pragma unroll
    for (int r = 0; r < 8; ++r) {
      rsum[r] += swz_xor<1>(rsum[r]);
      rsum[r] += swz_xor<2>(rsum[r]);
      rsum[r] += swz_xor<4>(rsum[r]);
      rsum[r] += swz_xor<8>(rsum[r]);
      lrow[r] = lrow[r] * alpha[r] + rsum[r];
    }

    // rescale running O
#pragma unroll
    for (int j = 0; j < 4; ++j)
#pragma unroll
      for (int r = 0; r < 8; ++r) oacc[j][r] *= alpha[r];

    // P: C-layout regs -> per-wave LDS -> A-layout fragment (in-wave, in-order)
#pragma unroll
    for (int st = 0; st < 2; ++st)
#pragma unroll
      for (int r = 0; r < 8; ++r)
        sP[w][8 * hi + r][st * 16 + l16] = (_Float16)s[st][r];
    v16h aP;
#pragma unroll
    for (int e = 0; e < 16; ++e) aP[e] = sP[w][l16][akk(e, hi)];

    // O += P * V : 4 N-subtiles of 16
    v16h bV[4];
#pragma unroll
    for (int j = 0; j < 4; ++j)
#pragma unroll
      for (int e = 0; e < 16; ++e) bV[j][e] = sV[bkk(e, hi)][16 * j + l16];
#pragma unroll
    for (int j = 0; j < 4; ++j)
      oacc[j] = __builtin_amdgcn_wmma_f32_16x16x32_f16(
          false, aP, false, bV[j], (short)0, oacc[j], false, false);
  }

  // normalize and write [B,T,H,hd] (== [B*T, C] row-major for out-projection)
  float rinv[8];
#pragma unroll
  for (int r = 0; r < 8; ++r) rinv[r] = 1.0f / lrow[r];
#pragma unroll
  for (int j = 0; j < 4; ++j)
#pragma unroll
    for (int r = 0; r < 8; ++r) {
      int qr = qbase + 16 * w + 8 * hi + r;
      float ov = oacc[j][r] * rinv[r];
      og[(((size_t)b * T_ + qr) * NH_ + h) * HD_ + 16 * j + l16] = (_Float16)ov;
    }
}

// ---------------------------------------------------------------------------
// Launch
// ---------------------------------------------------------------------------
extern "C" void kernel_launch(void* const* d_in, const int* in_sizes, int n_in,
                              void* d_out, int out_size, void* d_ws, size_t ws_size,
                              hipStream_t stream) {
  const float* x   = (const float*)d_in[0];
  const float* w_q = (const float*)d_in[1];
  const float* b_q = (const float*)d_in[2];
  const float* w_k = (const float*)d_in[3];
  const float* b_k = (const float*)d_in[4];
  const float* w_v = (const float*)d_in[5];
  const float* b_v = (const float*)d_in[6];
  const float* w_o = (const float*)d_in[7];
  const float* b_o = (const float*)d_in[8];
  float* out = (float*)d_out;

  char* ws = (char*)d_ws;
  const size_t szXH = (size_t)M_ * C_ * sizeof(_Float16);    // 8 MB
  const size_t szQ  = (size_t)M_ * C_ * sizeof(_Float16);    // 8 MB
  const size_t szK  = (size_t)M_ * KVC_ * sizeof(_Float16);  // 2 MB
  _Float16* xh = (_Float16*)(ws);
  _Float16* qh = (_Float16*)(ws + szXH);
  _Float16* kh = (_Float16*)(ws + szXH + szQ);
  _Float16* vh = (_Float16*)(ws + szXH + szQ + szK);
  _Float16* ah = (_Float16*)(ws + szXH + szQ + 2 * szK);

  // 1) x -> f16
  {
    int n = M_ * C_;
    cvt_f32_to_f16<<<(n + 255) / 256, 256, 0, stream>>>(x, xh, n);
  }
  // 2) Q/K/V projections (f16 out)
  gemm_wmma<false><<<dim3(C_ / 64, M_ / 256), 256, 0, stream>>>(
      xh, w_q, b_q, (void*)qh, M_, C_, C_);
  gemm_wmma<false><<<dim3(KVC_ / 64, M_ / 256), 256, 0, stream>>>(
      xh, w_k, b_k, (void*)kh, M_, KVC_, C_);
  gemm_wmma<false><<<dim3(KVC_ / 64, M_ / 256), 256, 0, stream>>>(
      xh, w_v, b_v, (void*)vh, M_, KVC_, C_);
  // 3) RoPE on Q and K
  {
    int nq = M_ * NH_ * 32;
    rope_kernel<<<(nq + 255) / 256, 256, 0, stream>>>(qh, NH_, nq);
    int nk = M_ * NKV_ * 32;
    rope_kernel<<<(nk + 255) / 256, 256, 0, stream>>>(kh, NKV_, nk);
  }
  // 4) causal flash attention
  flash_attn_kernel<<<dim3(T_ / 128, NH_, B_), 256, 0, stream>>>(qh, kh, vh, ah);
  // 5) output projection (f32 out + bias)
  gemm_wmma<true><<<dim3(C_ / 64, M_ / 256), 256, 0, stream>>>(
      ah, w_o, b_o, (void*)out, M_, C_, C_);
  (void)in_sizes; (void)n_in; (void)out_size; (void)ws_size;
}
